// GNNRefiner_16140487098441
// MI455X (gfx1250) — compile-verified, matrix operand order checked by reading, not run on hardware
//
#include <hip/hip_runtime.h>
#include <hip/hip_bf16.h>

typedef __attribute__((ext_vector_type(16))) _Float16 v16h;
typedef __attribute__((ext_vector_type(8)))  _Float16 v8h;
typedef __attribute__((ext_vector_type(8)))  float    v8f;

#define NPTS 8192
#define TD   384
#define KNN  16
#define KPAD 416           // 387 padded to multiple of 32 for 16x16x32 WMMA
#define NOUT 768           // [a | c]

// ---------------------------------------------------------------------------
// 1) KNN: per-thread query, LDS-tiled candidates, register top-16.
//    Key = |x_j|^2 - 2 x_i.x_j  (drops per-query constant |x_i|^2; rank-safe)
// ---------------------------------------------------------------------------
__global__ __launch_bounds__(256) void knn_kernel(const float* __restrict__ xyz,
                                                  int* __restrict__ knn_idx) {
  __shared__ float4 tile[256];
  const int q = blockIdx.x * 256 + threadIdx.x;
  const float qx = xyz[q * 3 + 0];
  const float qy = xyz[q * 3 + 1];
  const float qz = xyz[q * 3 + 2];

  float best[KNN];
  int   bidx[KNN];
#pragma unroll
  for (int s = 0; s < KNN; ++s) { best[s] = 3.4e38f; bidx[s] = 0; }

  for (int t0 = 0; t0 < NPTS; t0 += 256) {
    const int j = t0 + threadIdx.x;
    const float cx = xyz[j * 3 + 0];
    const float cy = xyz[j * 3 + 1];
    const float cz = xyz[j * 3 + 2];
    tile[threadIdx.x] = make_float4(cx, cy, cz, cx * cx + cy * cy + cz * cz);
    __syncthreads();
    for (int jj = 0; jj < 256; ++jj) {
      const float4 c = tile[jj];                 // LDS broadcast read
      const int cj = t0 + jj;
      const float dot = qx * c.x + qy * c.y + qz * c.z;
      const float key = fmaf(-2.0f, dot, c.w);
      if (cj != q && key < best[KNN - 1]) {      // rare: bubble insert
        float d = key; int id = cj;
#pragma unroll
        for (int s = 0; s < KNN; ++s) {
          if (d < best[s]) {
            float td = best[s]; int ti = bidx[s];
            best[s] = d; bidx[s] = id;
            d = td; id = ti;
          }
        }
      }
    }
    __syncthreads();
  }
#pragma unroll
  for (int s = 0; s < KNN; ++s) knn_idx[q * KNN + s] = bidx[s];
}

// ---------------------------------------------------------------------------
// 2) Casts: Xh[n][k] = f16( [feat | xyz | 0pad] ), WhT[f][k] = f16 transpose of
//    [W1_top | W1_bot] so B fragments load contiguously along K.
// ---------------------------------------------------------------------------
__global__ __launch_bounds__(256) void build_xh(const float* __restrict__ feat,
                                                const float* __restrict__ xyz,
                                                _Float16* __restrict__ Xh) {
  const int t = blockIdx.x * 256 + threadIdx.x;
  if (t >= NPTS * KPAD) return;
  const int n = t / KPAD, k = t % KPAD;
  float v = 0.0f;
  if (k < TD)          v = feat[n * TD + k];
  else if (k < TD + 3) v = xyz[n * 3 + (k - TD)];
  Xh[t] = (_Float16)v;
}

__global__ __launch_bounds__(256) void build_wht(const float* __restrict__ W1,
                                                 _Float16* __restrict__ WhT) {
  const int t = blockIdx.x * 256 + threadIdx.x;
  if (t >= NOUT * KPAD) return;
  const int f = t / KPAD, k = t % KPAD;
  float v = 0.0f;
  if (k < TD + 3)
    v = (f < TD) ? W1[k * TD + f]                        // W1_top column f
                 : W1[(TD + 3 + k) * TD + (f - TD)];     // W1_bot column f-384
  WhT[t] = (_Float16)v;
}

// ---------------------------------------------------------------------------
// 3) WMMA GEMM: AC(8192 x 768) = Xh(8192 x 416) * WhT^T, f16 in / f32 acc.
//    One wave per 16x16 C tile; block = 8 waves covering 16(M) x 128(N).
// ---------------------------------------------------------------------------
__global__ __launch_bounds__(256) void gemm_wmma(const _Float16* __restrict__ Xh,
                                                 const _Float16* __restrict__ WhT,
                                                 float* __restrict__ AC) {
  const int lane = threadIdx.x & 31;
  const int wave = threadIdx.x >> 5;
  const int m0 = blockIdx.x * 16;
  const int n0 = (blockIdx.y * 8 + wave) * 16;
  const int halfsel = lane >> 4;                    // K-half select per ISA layout
  const _Float16* aRow = Xh  + (size_t)(m0 + (lane & 15)) * KPAD + halfsel * 8;
  const _Float16* bRow = WhT + (size_t)(n0 + (lane & 15)) * KPAD + halfsel * 8;

  v8f acc = {};
  for (int k0 = 0; k0 < KPAD; k0 += 32) {
    const v8h a_lo = *(const v8h*)(aRow + k0);        // K = kb .. kb+7   (b128)
    const v8h a_hi = *(const v8h*)(aRow + k0 + 16);   // K = kb+16..kb+23 (b128)
    const v8h b_lo = *(const v8h*)(bRow + k0);
    const v8h b_hi = *(const v8h*)(bRow + k0 + 16);
    v16h a, b;
#pragma unroll
    for (int i = 0; i < 8; ++i) {
      a[i] = a_lo[i]; a[i + 8] = a_hi[i];
      b[i] = b_lo[i]; b[i + 8] = b_hi[i];
    }
    acc = __builtin_amdgcn_wmma_f32_16x16x32_f16(
        /*neg_a=*/false, a, /*neg_b=*/false, b,
        /*c_mod=*/(short)0, acc, /*reuse_a=*/false, /*reuse_b=*/false);
  }
  // C layout: VGPR r, lane l -> M = r + 8*(l>>4), N = l&15
#pragma unroll
  for (int r = 0; r < 8; ++r) {
    const int row = m0 + r + 8 * halfsel;
    const int col = n0 + (lane & 15);
    AC[(size_t)row * NOUT + col] = acc[r];
  }
}

// ---------------------------------------------------------------------------
// 4) Edge stage: block = 16 points x 16 neighbors.
//    Phase 1: u_p = a_p - c_p + b1 computed once into LDS (neighbor-invariant).
//    Phase 2: thread (p,k): m = relu(u_p + c_j) @ W2, all b128/float4;
//    cross-lane max over the 16 neighbor lanes (2 points per wave32), +b2, +xyz.
// ---------------------------------------------------------------------------
__global__ __launch_bounds__(256) void edge_kernel(const float* __restrict__ AC,
                                                   const int* __restrict__ knn_idx,
                                                   const float* __restrict__ xyz,
                                                   const float* __restrict__ b1,
                                                   const float* __restrict__ W2,
                                                   const float* __restrict__ b2,
                                                   float* __restrict__ out) {
  __shared__ float su[16][TD];          // 24 KB: u rows for this block's 16 points
  __shared__ float sW2x[TD], sW2y[TD], sW2z[TD];   // W2 column-major for f4 reads

  const int n_base = blockIdx.x * 16;

  for (int i = threadIdx.x; i < TD; i += 256) {
    sW2x[i] = W2[i * 3 + 0];
    sW2y[i] = W2[i * 3 + 1];
    sW2z[i] = W2[i * 3 + 2];
  }
  // u_p[d] = a_p[d] - c_p[d] + b1[d]; coalesced along d.
  for (int i = threadIdx.x; i < 16 * TD; i += 256) {
    const int p = i / TD, d = i % TD;
    const size_t row = (size_t)(n_base + p) * NOUT;
    su[p][d] = AC[row + d] - AC[row + TD + d] + b1[d];
  }
  __syncthreads();

  const int p = threadIdx.x >> 4;               // local point (lanes 0-15 / 16-31)
  const int k = threadIdx.x & 15;               // neighbor slot
  const int n = n_base + p;
  const int j = knn_idx[n * KNN + k];

  const float4* __restrict__ cj4 = (const float4*)(AC + (size_t)j * NOUT + TD);
  __builtin_prefetch(cj4, 0, 1);                // global_prefetch_b8
  const float4* __restrict__ up4 = (const float4*)(&su[p][0]);
  const float4* __restrict__ wx4 = (const float4*)sW2x;
  const float4* __restrict__ wy4 = (const float4*)sW2y;
  const float4* __restrict__ wz4 = (const float4*)sW2z;

  float a0 = 0.f, a1 = 0.f, a2 = 0.f;
#pragma unroll 2
  for (int d4 = 0; d4 < TD / 4; ++d4) {
    const float4 c = cj4[d4];                   // global b128 (L2 gather)
    const float4 u = up4[d4];                   // LDS b128 broadcast
    const float4 wx = wx4[d4], wy = wy4[d4], wz = wz4[d4];
    float t;
    t = u.x + c.x; t = t > 0.f ? t : 0.f;
    a0 = fmaf(t, wx.x, a0); a1 = fmaf(t, wy.x, a1); a2 = fmaf(t, wz.x, a2);
    t = u.y + c.y; t = t > 0.f ? t : 0.f;
    a0 = fmaf(t, wx.y, a0); a1 = fmaf(t, wy.y, a1); a2 = fmaf(t, wz.y, a2);
    t = u.z + c.z; t = t > 0.f ? t : 0.f;
    a0 = fmaf(t, wx.z, a0); a1 = fmaf(t, wy.z, a1); a2 = fmaf(t, wz.z, a2);
    t = u.w + c.w; t = t > 0.f ? t : 0.f;
    a0 = fmaf(t, wx.w, a0); a1 = fmaf(t, wy.w, a1); a2 = fmaf(t, wz.w, a2);
  }
  // max over 16 neighbor lanes (width-16 groups within the wave32)
#pragma unroll
  for (int m = 8; m >= 1; m >>= 1) {
    a0 = fmaxf(a0, __shfl_xor(a0, m, 16));
    a1 = fmaxf(a1, __shfl_xor(a1, m, 16));
    a2 = fmaxf(a2, __shfl_xor(a2, m, 16));
  }
  if (k == 0) {
    out[n * 3 + 0] = xyz[n * 3 + 0] + a0 + b2[0];
    out[n * 3 + 1] = xyz[n * 3 + 1] + a1 + b2[1];
    out[n * 3 + 2] = xyz[n * 3 + 2] + a2 + b2[2];
  }
}

// ---------------------------------------------------------------------------
extern "C" void kernel_launch(void* const* d_in, const int* in_sizes, int n_in,
                              void* d_out, int out_size, void* d_ws, size_t ws_size,
                              hipStream_t stream) {
  const float* xyz  = (const float*)d_in[0];
  const float* feat = (const float*)d_in[1];
  const float* W1   = (const float*)d_in[2];
  const float* b1   = (const float*)d_in[3];
  const float* W2   = (const float*)d_in[4];
  const float* b2   = (const float*)d_in[5];
  float* out = (float*)d_out;

  // Workspace carve (all offsets 16B aligned):
  char* ws = (char*)d_ws;
  _Float16* Xh  = (_Float16*)(ws);                       //  8192*416*2 = 6,815,744
  _Float16* WhT = (_Float16*)(ws + 6815744);             //   768*416*2 =   638,976
  float*    AC  = (float*)   (ws + 7454720);             //  8192*768*4 = 25,165,824
  int*      idx = (int*)     (ws + 32620544);            //  8192*16*4  =   524,288

  knn_kernel<<<NPTS / 256, 256, 0, stream>>>(xyz, idx);
  build_xh <<<(NPTS * KPAD) / 256, 256, 0, stream>>>(feat, xyz, Xh);
  build_wht<<<(NOUT * KPAD) / 256, 256, 0, stream>>>(W1, WhT);
  gemm_wmma<<<dim3(NPTS / 16, NOUT / 128), 256, 0, stream>>>(Xh, WhT, AC);
  edge_kernel<<<(NPTS * KNN) / 256, 256, 0, stream>>>(AC, idx, xyz, b1, W2, b2, out);
}
